// GraphSAGE_19808389169944
// MI455X (gfx1250) — compile-verified, matrix operand order checked by reading, not run on hardware
//
#include <hip/hip_runtime.h>
#include <hip/hip_bf16.h>

#define N_NODES 50000
#define N_EDGES 800000
#define DIM_IN  128
#define DIM_H   128
#define NTYPES  3

typedef __attribute__((ext_vector_type(16))) __bf16 v16bf;
typedef __attribute__((ext_vector_type(8)))  float  v8f;

// ---------------------------------------------------------------- utilities
__global__ void zero_f32(float* __restrict__ p, int n) {
    int i = blockIdx.x * blockDim.x + threadIdx.x;
    if (i < n) p[i] = 0.0f;
}

// per-(type,dst) edge counts; identical for both layers (depends only on graph)
__global__ void count_edges(const int* __restrict__ dst,
                            const int* __restrict__ ew,
                            float* __restrict__ cnt, int nE, int nN) {
    int e = blockIdx.x * blockDim.x + threadIdx.x;
    if (e < nE) atomicAdd(&cnt[ew[e] * nN + dst[e]], 1.0f);
}

// Pre-pack weight matrix W[128][K] (fp32, row-major [n][k]) into WMMA B-fragment
// order, bf16:  frag f = (ksIdx*8 + nt)*32 + lane, 16 bf16 per frag (32B, contiguous).
// lane: laneLo=n%16, laneHi selects K sub-range {0..7,16..23} / {8..15,24..31}.
__global__ void pack_w(const float* __restrict__ W, __bf16* __restrict__ out, int K) {
    int idx   = blockIdx.x * blockDim.x + threadIdx.x;
    int total = (K / 32) * 8 * 32;
    if (idx >= total) return;
    int lane = idx & 31;
    int nt   = (idx >> 5) & 7;
    int ks   = idx >> 8;
    int laneLo = lane & 15, laneHi = lane >> 4;
    const float* row = W + (size_t)(nt * 16 + laneLo) * K + ks * 32 + laneHi * 8;
    __bf16* o = out + (size_t)idx * 16;
#pragma unroll
    for (int i = 0; i < 8; ++i) {
        o[i]     = (__bf16)row[i];
        o[i + 8] = (__bf16)row[i + 16];
    }
}

// scatter-add feature rows of edges of type t: one wave per edge (uniform branch)
template <int D>
__global__ __launch_bounds__(256)
void scatter_edges(const int* __restrict__ src, const int* __restrict__ dst,
                   const int* __restrict__ ew, const float* __restrict__ X,
                   float* __restrict__ ssum, int t, int nE) {
    int wave   = (blockIdx.x * blockDim.x + threadIdx.x) >> 5;
    int lane   = threadIdx.x & 31;
    int nWaves = (gridDim.x * blockDim.x) >> 5;
    for (int e = wave; e < nE; e += nWaves) {
        if (ew[e] != t) continue;                 // wave-uniform
        const float* row  = X    + (size_t)src[e] * D;
        float*       orow = ssum + (size_t)dst[e] * D;
#pragma unroll
        for (int j = 0; j < D / 128; ++j) {
            int d = lane * 4 + j * 128;
            float4 v = *(const float4*)(row + d);
            atomicAdd(orow + d + 0, v.x);
            atomicAdd(orow + d + 1, v.y);
            atomicAdd(orow + d + 2, v.z);
            atomicAdd(orow + d + 3, v.w);
        }
    }
}

// ---------------------------------------------------------------- WMMA GEMM
// A-fragment slice for this lane from an fp32 row, scaled then cvt to bf16.
// Loads issued as four independent float4s (distinct regs -> one clause, one wait),
// then scaled/converted.  Layout: bf16[0..7] = K {0..7}, bf16[8..15] = K {16..23}
// (pre-offset by 8*laneHi selects the lane-high sub-ranges).
__device__ __forceinline__ v16bf load_afrag(const float* __restrict__ p, float s) {
    float4 x0 = *(const float4*)(p + 0);
    float4 x1 = *(const float4*)(p + 4);
    float4 x2 = *(const float4*)(p + 16);
    float4 x3 = *(const float4*)(p + 20);
    v16bf v;
    v[0]  = (__bf16)(x0.x * s); v[1]  = (__bf16)(x0.y * s);
    v[2]  = (__bf16)(x0.z * s); v[3]  = (__bf16)(x0.w * s);
    v[4]  = (__bf16)(x1.x * s); v[5]  = (__bf16)(x1.y * s);
    v[6]  = (__bf16)(x1.z * s); v[7]  = (__bf16)(x1.w * s);
    v[8]  = (__bf16)(x2.x * s); v[9]  = (__bf16)(x2.y * s);
    v[10] = (__bf16)(x2.z * s); v[11] = (__bf16)(x2.w * s);
    v[12] = (__bf16)(x3.x * s); v[13] = (__bf16)(x3.y * s);
    v[14] = (__bf16)(x3.z * s); v[15] = (__bf16)(x3.w * s);
    return v;
}

// out[:, colBase:colBase+128] = relu( (ssum/max(cnt,1)) @ Wl^T + X @ Wr^T + b )
// Block: 256 threads = 8 waves, tile 128 rows x 128 cols; wave w -> rows [w*16, w*16+16)
// Wlp/Wrp are fragment-packed bf16 (see pack_w): B loads are contiguous 32B/lane.
template <int K>
__global__ __launch_bounds__(256)
void sage_gemm(const float* __restrict__ ssum, const float* __restrict__ X,
               const __bf16* __restrict__ Wlp, const __bf16* __restrict__ Wrp,
               const float* __restrict__ bias, const float* __restrict__ cnt,
               float* __restrict__ out, int outStride, int colBase, int M) {
    const int wave    = threadIdx.x >> 5;
    const int lane    = threadIdx.x & 31;
    const int laneLo  = lane & 15;
    const int laneHi  = lane >> 4;
    const int rowBase = blockIdx.x * 128 + wave * 16;
    int row  = rowBase + laneLo;
    int rowc = row < M ? row : (M - 1);           // clamp (stores are masked)

    const float inv = 1.0f / fmaxf(cnt[rowc], 1.0f);
    const float* a1 = ssum + (size_t)rowc * K + laneHi * 8;  // mean operand
    const float* a2 = X    + (size_t)rowc * K + laneHi * 8;  // self operand
    const v16bf* Bl = (const v16bf*)Wlp;
    const v16bf* Br = (const v16bf*)Wrp;

    __builtin_prefetch(a2, 0, 0);                 // warm pass-2 row (global_prefetch_b8)

    v8f acc[8];
#pragma unroll
    for (int nt = 0; nt < 8; ++nt)
        acc[nt] = (v8f){0.f, 0.f, 0.f, 0.f, 0.f, 0.f, 0.f, 0.f};

    // pass 1: mean @ Wl^T   (mean folded into A-load via `inv`)
#pragma unroll
    for (int ki = 0; ki < K / 32; ++ki) {
        v16bf a = load_afrag(a1 + ki * 32, inv);
#pragma unroll
        for (int nt = 0; nt < 8; ++nt) {
            v16bf b = Bl[(ki * 8 + nt) * 32 + lane];
            acc[nt] = __builtin_amdgcn_wmma_f32_16x16x32_bf16(
                false, a, false, b, (short)0, acc[nt], false, false);
        }
    }
    // pass 2: X @ Wr^T
#pragma unroll
    for (int ki = 0; ki < K / 32; ++ki) {
        v16bf a = load_afrag(a2 + ki * 32, 1.0f);
#pragma unroll
        for (int nt = 0; nt < 8; ++nt) {
            v16bf b = Br[(ki * 8 + nt) * 32 + lane];
            acc[nt] = __builtin_amdgcn_wmma_f32_16x16x32_bf16(
                false, a, false, b, (short)0, acc[nt], false, false);
        }
    }

    // bias + relu + store.  D layout: vgpr r -> M = r + 8*laneHi, N = laneLo
#pragma unroll
    for (int nt = 0; nt < 8; ++nt) {
        const float bv  = bias[nt * 16 + laneLo];
        const int   col = colBase + nt * 16 + laneLo;
#pragma unroll
        for (int r = 0; r < 8; ++r) {
            int m = rowBase + r + laneHi * 8;
            if (m < M) out[(size_t)m * outStride + col] = fmaxf(acc[nt][r] + bv, 0.0f);
        }
    }
}

// ---------------------------------------------------------------- launcher
extern "C" void kernel_launch(void* const* d_in, const int* in_sizes, int n_in,
                              void* d_out, int out_size, void* d_ws, size_t ws_size,
                              hipStream_t stream) {
    const float* x   = (const float*)d_in[0];
    const int*   ei  = (const int*)  d_in[1];
    const int*   ew  = (const int*)  d_in[2];
    const float* Wl0 = (const float*)d_in[3];
    const float* Wr0 = (const float*)d_in[4];
    const float* bl0 = (const float*)d_in[5];
    const float* Wl1 = (const float*)d_in[6];
    const float* Wr1 = (const float*)d_in[7];
    const float* bl1 = (const float*)d_in[8];
    float* out = (float*)d_out;

    float* ws   = (float*)d_ws;
    float* cnt  = ws;                               // 3*50000 floats
    float* ssum = ws + 150016;                      // 50000*384 floats (reused per type)
    float* h    = ssum + (size_t)N_NODES * 384;     // 50000*384 floats (layer-0 output)
    __bf16* wb  = (__bf16*)(h + (size_t)N_NODES * 384);  // packed weights, 32B-aligned

    const int PK0 = DIM_H * DIM_IN;                 // 16384 elems per packed 128-K matrix
    const int PK1 = DIM_H * DIM_H * NTYPES;         // 49152 elems per packed 384-K matrix
    __bf16* Wl0p = wb;                              // [3][PK0]
    __bf16* Wr0p = Wl0p + (size_t)NTYPES * PK0;
    __bf16* Wl1p = Wr0p + (size_t)NTYPES * PK0;     // [3][PK1]
    __bf16* Wr1p = Wl1p + (size_t)NTYPES * PK1;

    const int* src = ei;             // edge_index[0]
    const int* dst = ei + N_EDGES;   // edge_index[1]

    const int gemmBlocks = (N_NODES + 127) / 128;   // 391

    // pack all weights to fragment-major bf16 (once per launch; tiny)
    for (int t = 0; t < NTYPES; ++t) {
        pack_w<<<(PK0 / 16 + 255) / 256, 256, 0, stream>>>(
            Wl0 + (size_t)t * PK0, Wl0p + (size_t)t * PK0, DIM_IN);
        pack_w<<<(PK0 / 16 + 255) / 256, 256, 0, stream>>>(
            Wr0 + (size_t)t * PK0, Wr0p + (size_t)t * PK0, DIM_IN);
        pack_w<<<(PK1 / 16 + 255) / 256, 256, 0, stream>>>(
            Wl1 + (size_t)t * PK1, Wl1p + (size_t)t * PK1, DIM_H * NTYPES);
        pack_w<<<(PK1 / 16 + 255) / 256, 256, 0, stream>>>(
            Wr1 + (size_t)t * PK1, Wr1p + (size_t)t * PK1, DIM_H * NTYPES);
    }

    // edge-type counts (shared by both layers)
    zero_f32<<<(NTYPES * N_NODES + 255) / 256, 256, 0, stream>>>(cnt, NTYPES * N_NODES);
    count_edges<<<(N_EDGES + 255) / 256, 256, 0, stream>>>(dst, ew, cnt, N_EDGES, N_NODES);

    // ---- layer 0: x [50000,128] -> h [50000,384]
    for (int t = 0; t < NTYPES; ++t) {
        zero_f32<<<(N_NODES * DIM_IN + 255) / 256, 256, 0, stream>>>(ssum, N_NODES * DIM_IN);
        scatter_edges<DIM_IN><<<4096, 256, 0, stream>>>(src, dst, ew, x, ssum, t, N_EDGES);
        sage_gemm<DIM_IN><<<gemmBlocks, 256, 0, stream>>>(
            ssum, x,
            Wl0p + (size_t)t * PK0, Wr0p + (size_t)t * PK0,
            bl0 + t * DIM_H, cnt + (size_t)t * N_NODES,
            h, DIM_H * NTYPES, t * DIM_H, N_NODES);
    }

    // ---- layer 1: h [50000,384] -> out [50000,384]
    for (int t = 0; t < NTYPES; ++t) {
        zero_f32<<<(N_NODES * 384 + 255) / 256, 256, 0, stream>>>(ssum, N_NODES * 384);
        scatter_edges<384><<<4096, 256, 0, stream>>>(src, dst, ew, h, ssum, t, N_EDGES);
        sage_gemm<384><<<gemmBlocks, 256, 0, stream>>>(
            ssum, h,
            Wl1p + (size_t)t * PK1, Wr1p + (size_t)t * PK1,
            bl1 + t * DIM_H, cnt + (size_t)t * N_NODES,
            out, 384, t * DIM_H, N_NODES);
    }
}
